// MemoryModule_12146167513159
// MI455X (gfx1250) — matrix operand order repaired and checked by reading, not compiled, not run on hardware
//
#include <hip/hip_runtime.h>

typedef __attribute__((ext_vector_type(16))) __bf16 v16bf;
typedef __attribute__((ext_vector_type(8)))  float  v8f;
typedef __attribute__((ext_vector_type(4)))  unsigned int u32x4;

#define NFACT   262144
#define HDIM    100
#define KPAD    224          // 7 * 32
#define NPAD    112          // 7 * 16
#define NFRAG   49           // 7 kc * 7 nt
#define FACTS_PER_BLOCK 128  // 8 waves * 16 facts
#define K1_BLOCKS (NFACT / FACTS_PER_BLOCK)   // 2048
#define K3_BLOCKS (NFACT / 256)               // 1024

__device__ __forceinline__ unsigned short f2bf(float f) {
    unsigned u = __builtin_bit_cast(unsigned, f);
    unsigned r = u + 0x7FFFu + ((u >> 16) & 1u);   // round-to-nearest-even
    return (unsigned short)(r >> 16);
}

union FragU { u32x4 q[2]; v16bf v; };

// A-fragment: elements 0..7 at p, elements 8..15 at p+16 (K stride 16)
__device__ __forceinline__ v16bf load_afrag(const unsigned short* p) {
    FragU u;
    u.q[0] = *reinterpret_cast<const u32x4*>(p);
    u.q[1] = *reinterpret_cast<const u32x4*>(p + 16);
    return u.v;
}
// B-fragment: 16 contiguous bf16 per lane (pre-swizzled)
__device__ __forceinline__ v16bf load_bfrag(const unsigned short* p) {
    FragU u;
    u.q[0] = *reinterpret_cast<const u32x4*>(p);
    u.q[1] = *reinterpret_cast<const u32x4*>(p + 8);
    return u.v;
}

// ---------------- k0: W1 -> bf16 B-fragments in workspace ----------------
// B layout (32x16 bf16): h=lane/16, elem i<8 -> K=8h+i, elem i>=8 -> K=16+8h+(i-8); N = lane%16
__global__ void k0_prep_w1(const float* __restrict__ W1, unsigned short* __restrict__ wfrag) {
    int idx = blockIdx.x * blockDim.x + threadIdx.x;
    if (idx >= NFRAG * 512) return;
    int frag = idx >> 9;
    int r    = idx & 511;
    int lane = r >> 4;
    int i    = r & 15;
    int kc = frag / 7, nt = frag % 7;
    int k = kc * 32 + ((lane & 16) ? 8 : 0) + ((i < 8) ? i : 16 + (i - 8));
    int n = nt * 16 + (lane & 15);
    float v = (k < 2 * HDIM && n < HDIM) ? W1[n * (2 * HDIM) + k] : 0.0f;
    wfrag[idx] = f2bf(v);
}

// ---------------- k1: scores via bf16 WMMA ----------------
__global__ __launch_bounds__(256)
void k1_scores(const float* __restrict__ f, const float* __restrict__ m,
               const float* __restrict__ oc, const float* __restrict__ b1,
               const float* __restrict__ W2, const float* __restrict__ b2p,
               const unsigned short* __restrict__ wfragG,
               float* __restrict__ scores, float* __restrict__ blockmax) {
    __shared__ __align__(16) unsigned short sWf[NFRAG * 512];   // 50176 B
    __shared__ __align__(16) unsigned short sZ[8 * 16 * KPAD];  // 57344 B
    __shared__ float sOC[HDIM], sM[HDIM], sB1[NPAD], sW2[NPAD];
    __shared__ float sRed[256];

    const int t = threadIdx.x;

    // stage pre-swizzled W1 fragments ws -> LDS (wide 16B copies)
    {
        const u32x4* src = reinterpret_cast<const u32x4*>(wfragG);
        u32x4* dst = reinterpret_cast<u32x4*>(sWf);
        for (int i = t; i < (NFRAG * 512) / 8; i += 256) dst[i] = src[i];
    }
    if (t < HDIM) { sOC[t] = oc[t]; sM[t] = m[t]; }
    if (t < NPAD) {
        sB1[t] = (t < HDIM) ? b1[t] : 0.0f;
        sW2[t] = (t < HDIM) ? W2[t] : 0.0f;
    }
    __syncthreads();

    // stage z = [|f-oc| ; |f-m|] as bf16, zero-padded to KPAD
    const long tile0 = (long)blockIdx.x * FACTS_PER_BLOCK;
    for (int i = t; i < 8 * 16 * KPAD; i += 256) {
        int k   = i % KPAD;
        int row = i / KPAD;                      // 0..127
        float v = 0.0f;
        if (k < HDIM)            v = fabsf(f[(tile0 + row) * HDIM + k] - sOC[k]);
        else if (k < 2 * HDIM)   v = fabsf(f[(tile0 + row) * HDIM + (k - HDIM)] - sM[k - HDIM]);
        sZ[i] = f2bf(v);
    }
    __syncthreads();

    // per-wave WMMA: 16 facts x 112 features, K = 224 in 7 chunks of 32
    const int wv   = t >> 5;
    const int lane = t & 31;
    const int hs   = lane >> 4;      // K-half select for A
    const int Mr   = lane & 15;      // A row
    const unsigned short* zw = sZ + wv * 16 * KPAD;

    v8f acc[7];
    #pragma unroll
    for (int nt = 0; nt < 7; ++nt) acc[nt] = v8f{};

    #pragma unroll
    for (int kc = 0; kc < 7; ++kc) {
        v16bf afrag = load_afrag(zw + Mr * KPAD + kc * 32 + hs * 8);
        #pragma unroll
        for (int nt = 0; nt < 7; ++nt) {
            v16bf bfrag = load_bfrag(sWf + (kc * 7 + nt) * 512 + lane * 16);
            acc[nt] = __builtin_amdgcn_wmma_f32_16x16x32_bf16(
                false, afrag, false, bfrag, (short)0, acc[nt], false, false);
        }
    }

    // scores: s[M] = sum_n tanh(acc + b1[n]) * W2[n] + b2
    // C layout: vgpr v <-> M = v + 8*hs ; N = nt*16 + lane%16
    const float b2 = b2p[0];
    const int nl = lane & 15;
    float ps[8];
    #pragma unroll
    for (int v = 0; v < 8; ++v) ps[v] = 0.0f;
    #pragma unroll
    for (int nt = 0; nt < 7; ++nt) {
        int n = nt * 16 + nl;
        float w2 = sW2[n], bb = sB1[n];  // zero-padded -> no branch needed
        #pragma unroll
        for (int v = 0; v < 8; ++v) ps[v] += tanhf(acc[nt][v] + bb) * w2;
    }
    // butterfly reduce across the 16-lane group (wave32)
    #pragma unroll
    for (int v = 0; v < 8; ++v)
        for (int mk = 1; mk < 16; mk <<= 1) ps[v] += __shfl_xor(ps[v], mk, 32);

    const long tb = tile0 + wv * 16 + hs * 8;
    float lmax = -3.4e38f;
    #pragma unroll
    for (int v = 0; v < 8; ++v) {
        float s = ps[v] + b2;
        lmax = fmaxf(lmax, s);
        if (nl == 0) scores[tb + v] = s;   // lane 0 and lane 16 each write their 8
    }

    // block max -> blockmax[blockIdx]
    sRed[t] = lmax;
    __syncthreads();
    for (int s = 128; s > 0; s >>= 1) {
        if (t < s) sRed[t] = fmaxf(sRed[t], sRed[t + s]);
        __syncthreads();
    }
    if (t == 0) blockmax[blockIdx.x] = sRed[0];
}

// ---------------- k2: global max ----------------
__global__ void k2_gmax(const float* __restrict__ blockmax, float* __restrict__ gmax) {
    __shared__ float red[256];
    int t = threadIdx.x;
    float mx = -3.4e38f;
    for (int i = t; i < K1_BLOCKS; i += 256) mx = fmaxf(mx, blockmax[i]);
    red[t] = mx;
    __syncthreads();
    for (int s = 128; s > 0; s >>= 1) {
        if (t < s) red[t] = fmaxf(red[t], red[t + s]);
        __syncthreads();
    }
    if (t == 0) *gmax = red[0];
}

// ---------------- k3: exp weights + partial weighted sums (deterministic) --------
__global__ __launch_bounds__(256)
void k3_weighted(const float* __restrict__ scores, const float* __restrict__ f,
                 const float* __restrict__ gmaxp,
                 float* __restrict__ bsum, float* __restrict__ cpart) {
    __shared__ float wbuf[256];
    __shared__ float red[256];
    __shared__ float cp[256];
    const int t = threadIdx.x;
    const long base = (long)blockIdx.x * 256;
    const float gmax = *gmaxp;

    float w = __expf(scores[base + t] - gmax);
    wbuf[t] = w;
    red[t] = w;
    __syncthreads();
    for (int s = 128; s > 0; s >>= 1) {
        if (t < s) red[t] += red[t + s];
        __syncthreads();
    }
    if (t == 0) bsum[blockIdx.x] = red[0];

    const int h = t & 127, fg = t >> 7;
    float c = 0.0f;
    if (h < HDIM) {
        const float* fp = f + base * HDIM + h;
        for (int n = fg; n < 256; n += 2) c += wbuf[n] * fp[(long)n * HDIM];
    }
    cp[t] = c;
    __syncthreads();
    if (t < HDIM) cpart[(long)blockIdx.x * HDIM + t] = cp[t] + cp[128 + t];
}

// ---------------- k4: final reduce + gated update ----------------
__global__ void k4_final(const float* __restrict__ m, const float* __restrict__ oc,
                         const float* __restrict__ bsum, const float* __restrict__ cpart,
                         const float* __restrict__ W3, const float* __restrict__ b3,
                         float* __restrict__ out) {
    __shared__ float red[256];
    __shared__ float cat[3 * HDIM];
    const int t = threadIdx.x;  // 256

    float s = 0.0f;
    for (int i = t; i < K3_BLOCKS; i += 256) s += bsum[i];
    red[t] = s;
    __syncthreads();
    for (int k = 128; k > 0; k >>= 1) {
        if (t < k) red[t] += red[t + k];
        __syncthreads();
    }
    const float gsum = red[0];

    if (t < HDIM) {
        float c = 0.0f;
        for (int b = 0; b < K3_BLOCKS; ++b) c += cpart[(long)b * HDIM + t];
        cat[t]            = m[t];
        cat[HDIM + t]     = c / gsum;
        cat[2 * HDIM + t] = oc[t];
    }
    __syncthreads();
    if (t < HDIM) {
        float a = b3[t];
        const float* w = W3 + t * (3 * HDIM);
        for (int i = 0; i < 3 * HDIM; ++i) a += w[i] * cat[i];
        out[t] = fmaxf(a, 0.0f);
    }
}

extern "C" void kernel_launch(void* const* d_in, const int* in_sizes, int n_in,
                              void* d_out, int out_size, void* d_ws, size_t ws_size,
                              hipStream_t stream) {
    const float* m  = (const float*)d_in[0];
    const float* oh = (const float*)d_in[1];   // output_hidden [1,1,H] -> oc
    const float* f  = (const float*)d_in[2];
    const float* W1 = (const float*)d_in[3];
    const float* b1 = (const float*)d_in[4];
    const float* W2 = (const float*)d_in[5];
    const float* b2 = (const float*)d_in[6];
    const float* W3 = (const float*)d_in[7];
    const float* b3 = (const float*)d_in[8];
    float* out = (float*)d_out;

    char* ws = (char*)d_ws;
    float*          scores   = (float*)(ws);                       // N floats
    unsigned short* wfrag    = (unsigned short*)(ws + 1048576);    // 50176 B
    float*          blockmax = (float*)(ws + 1098752);             // 2048 floats
    float*          bsum     = (float*)(ws + 1106944);             // 1024 floats
    float*          cpart    = (float*)(ws + 1111040);             // 1024*100 floats
    float*          gmax     = (float*)(ws + 1520640);             // 1 float

    k0_prep_w1<<<(NFRAG * 512 + 255) / 256, 256, 0, stream>>>(W1, wfrag);
    k1_scores<<<K1_BLOCKS, 256, 0, stream>>>(f, m, oh, b1, W2, b2, wfrag, scores, blockmax);
    k2_gmax<<<1, 256, 0, stream>>>(blockmax, gmax);
    k3_weighted<<<K3_BLOCKS, 256, 0, stream>>>(scores, f, gmax, bsum, cpart);
    k4_final<<<1, 256, 0, stream>>>(m, oh, bsum, cpart, W3, b3, out);
}
